// YalmSelfAttention_5205500363030
// MI455X (gfx1250) — matrix operand order repaired and checked by reading, not compiled, not run on hardware
//
#include <hip/hip_runtime.h>
#include <hip/hip_bf16.h>
#include <math.h>

typedef __attribute__((ext_vector_type(16))) _Float16 v16h;
typedef __attribute__((ext_vector_type(8)))  _Float16 v8h;
typedef __attribute__((ext_vector_type(4)))  _Float16 v4h;
typedef __attribute__((ext_vector_type(2)))  _Float16 v2h;
typedef __attribute__((ext_vector_type(8)))  float    v8f;
typedef __attribute__((ext_vector_type(4)))  int      v4i;

#define S_LEN 2048
#define BATCH 2
#define HID   2048
#define NHEAD 16
#define HN    128
#define MROWS (S_LEN * BATCH)   // 4096

#if defined(__has_builtin)
#  if __has_builtin(__builtin_amdgcn_global_load_async_to_lds_b128)
#    define HAVE_ASYNC_LDS 1
#  endif
#endif

static __device__ __forceinline__ v16h cat8(v8h lo, v8h hi) {
  return __builtin_shufflevector(lo, hi, 0,1,2,3,4,5,6,7,8,9,10,11,12,13,14,15);
}
static __device__ __forceinline__ v8f wmma16(v16h a, v16h b, v8f c) {
  return __builtin_amdgcn_wmma_f32_16x16x32_f16(false, a, false, b, (short)0, c, false, false);
}

#ifdef HAVE_ASYNC_LDS
typedef v4i __attribute__((address_space(1)))* gv4i_p;   // global int4*
typedef v4i __attribute__((address_space(3)))* lv4i_p;   // LDS int4*
static __device__ __forceinline__ void async_cp16(const _Float16* g, _Float16* l) {
  gv4i_p gp = (gv4i_p)(unsigned long long)g;
  lv4i_p lp = (lv4i_p)(unsigned int)(unsigned long long)l;
  __builtin_amdgcn_global_load_async_to_lds_b128(gp, lp, 0, 0);
}
static __device__ __forceinline__ void wait_async0() {
#if __has_builtin(__builtin_amdgcn_s_wait_asynccnt)
  __builtin_amdgcn_s_wait_asynccnt(0);
#else
  asm volatile("s_wait_asynccnt 0" ::: "memory");
#endif
}
#endif

// ---------------------------------------------------------------------------
// GEMM: C[M,N] = A[M,K] * W[N,K]^T + bias[N]   (fp32 in/out, f16 WMMA core)
// 256 threads / 8 waves; block tile 256x128 (waves 4Mx2N, 64x64 each);
// K-step 32; double-buffered LDS with register prefetch.
// ---------------------------------------------------------------------------
#define GLDT 40   // LDS f16 row stride (80B = 5*16B -> aligned b128, conflict-free)

static __device__ __forceinline__ void load_tileA(const float* A, int K, int bm,
                                                  int k0, int tid, float4 pa[8]) {
#pragma unroll
  for (int i = 0; i < 8; ++i) {
    int idx = tid + i * 256;            // 2048 float4 = 256 rows x 32 cols
    int r = idx >> 3, c = (idx & 7) << 2;
    pa[i] = *(const float4*)(A + (size_t)(bm + r) * K + k0 + c);
  }
}
static __device__ __forceinline__ void load_tileB(const float* W, int K, int bn,
                                                  int k0, int tid, float4 pb[4]) {
#pragma unroll
  for (int i = 0; i < 4; ++i) {
    int idx = tid + i * 256;            // 1024 float4 = 128 rows x 32 cols
    int r = idx >> 3, c = (idx & 7) << 2;
    pb[i] = *(const float4*)(W + (size_t)(bn + r) * K + k0 + c);
  }
}
static __device__ __forceinline__ void cvt_store(_Float16* dst, int r, int c, float4 a4) {
  v4h h; h.x = (_Float16)a4.x; h.y = (_Float16)a4.y;
         h.z = (_Float16)a4.z; h.w = (_Float16)a4.w;
  *(v4h*)&dst[r * GLDT + c] = h;
}
static __device__ __forceinline__ void store_tileA(_Float16* As, int tid, const float4 pa[8]) {
#pragma unroll
  for (int i = 0; i < 8; ++i) {
    int idx = tid + i * 256;
    cvt_store(As, idx >> 3, (idx & 7) << 2, pa[i]);
  }
}
static __device__ __forceinline__ void store_tileB(_Float16* Bs, int tid, const float4 pb[4]) {
#pragma unroll
  for (int i = 0; i < 4; ++i) {
    int idx = tid + i * 256;
    cvt_store(Bs, idx >> 3, (idx & 7) << 2, pb[i]);
  }
}

__global__ __launch_bounds__(256) void gemm_bias_wmma(
    const float* __restrict__ A, const float* __restrict__ W,
    const float* __restrict__ bias, float* __restrict__ C,
    int M, int N, int K)
{
  __shared__ __align__(16) _Float16 As[2][256 * GLDT];
  __shared__ __align__(16) _Float16 Bs[2][128 * GLDT];

  const int tid  = threadIdx.x;
  const int lane = tid & 31;
  const int wave = tid >> 5;
  const int hlf  = lane >> 4;
  const int l16  = lane & 15;
  const int bm   = blockIdx.y * 256;
  const int bn   = blockIdx.x * 128;
  const int wm0  = (wave >> 1) * 64;   // 4 waves over M
  const int wn0  = (wave & 1) * 64;    // 2 waves over N

  v8f zero = {0,0,0,0,0,0,0,0};
  v8f acc[4][4];
#pragma unroll
  for (int i = 0; i < 4; ++i)
#pragma unroll
    for (int j = 0; j < 4; ++j) acc[i][j] = zero;

  float4 pa[8], pb[4];
  load_tileA(A, K, bm, 0, tid, pa);
  load_tileB(W, K, bn, 0, tid, pb);
  store_tileA(As[0], tid, pa);
  store_tileB(Bs[0], tid, pb);
  __syncthreads();

  int cur = 0;
  const int nk = K >> 5;
  for (int kb = 0; kb < nk; ++kb) {
    const bool more = (kb + 1) < nk;
    if (more) {
      load_tileA(A, K, bm, (kb + 1) * 32, tid, pa);
      load_tileB(W, K, bn, (kb + 1) * 32, tid, pb);
    }
    // B fragments: lane = column n, K-half = lane/16, 16 contiguous K per lane
    v16h bf[4];
#pragma unroll
    for (int fn = 0; fn < 4; ++fn) {
      const _Float16* p = &Bs[cur][(wn0 + fn * 16 + l16) * GLDT + hlf * 16];
      bf[fn] = cat8(*(const v8h*)(p), *(const v8h*)(p + 8));
    }
    // A fragments: lane-half h holds K = {16h..16h+15}
#pragma unroll
    for (int fm = 0; fm < 4; ++fm) {
      const _Float16* p = &As[cur][(wm0 + fm * 16 + l16) * GLDT];
      v16h af = cat8(*(const v8h*)(p + hlf * 16), *(const v8h*)(p + hlf * 16 + 8));
#pragma unroll
      for (int fn = 0; fn < 4; ++fn)
        acc[fm][fn] = wmma16(af, bf[fn], acc[fm][fn]);
    }
    if (more) {
      store_tileA(As[cur ^ 1], tid, pa);
      store_tileB(Bs[cur ^ 1], tid, pb);
      cur ^= 1;
      __syncthreads();
    }
  }

  // Epilogue: bias + store (C layout: row = r + 8*half, col = lane%16)
  float bv[4];
#pragma unroll
  for (int fn = 0; fn < 4; ++fn) bv[fn] = bias[bn + wn0 + fn * 16 + l16];
#pragma unroll
  for (int fm = 0; fm < 4; ++fm) {
#pragma unroll
    for (int r = 0; r < 8; ++r) {
      int row = bm + wm0 + fm * 16 + r + 8 * hlf;
#pragma unroll
      for (int fn = 0; fn < 4; ++fn) {
        int col = bn + wn0 + fn * 16 + l16;
        C[(size_t)row * N + col] = acc[fm][fn][r] + bv[fn];
      }
    }
  }
}

// ---------------------------------------------------------------------------
// RoPE + QKV split: mixed fp32 (S,B,NH,3*HN) -> q,k (rotated) and v, f16,
// layout [b][n][s][hn] for the attention kernel.
// ---------------------------------------------------------------------------
__global__ __launch_bounds__(256) void rope_split(
    const float* __restrict__ mixed,
    _Float16* __restrict__ q, _Float16* __restrict__ k, _Float16* __restrict__ v)
{
  int t = blockIdx.x * blockDim.x + threadIdx.x;   // S*B*NH*64 threads
  int i = t & 63;
  int n = (t >> 6) & (NHEAD - 1);
  int b = (t >> 10) & (BATCH - 1);
  int s = t >> 11;

  const float* base = mixed + ((size_t)(s * BATCH + b)) * (3 * HID) + n * (3 * HN);
  float ang = (float)s * __expf((float)i * -0.14391157f);  // 1e-4^(i/64)
  float cs, sn;
  __sincosf(ang, &cs, &sn);

  float q0 = base[i],        q1 = base[i + 64];
  float k0 = base[HN + i],   k1 = base[HN + 64 + i];
  float v0 = base[2*HN + i], v1 = base[2*HN + 64 + i];

  size_t o = ((size_t)((b * NHEAD + n) * S_LEN + s)) * HN + i;
  q[o]      = (_Float16)(cs * q0 - sn * q1);
  q[o + 64] = (_Float16)(sn * q0 + cs * q1);
  k[o]      = (_Float16)(cs * k0 - sn * k1);
  k[o + 64] = (_Float16)(sn * k0 + cs * k1);
  v[o]      = (_Float16)v0;
  v[o + 64] = (_Float16)v1;
}

// ---------------------------------------------------------------------------
// Flash attention: one workgroup per (b,n, 128-query block); 8 waves, each
// owning 16 query rows. Online softmax, analytic causal mask, double-buffered
// K/V tiles (K staged via async global->LDS when available).
// ---------------------------------------------------------------------------
#define KLD 136   // Ks row stride f16 (272B = 17*16B)
#define VLD 40    // Vs (transposed) row stride f16 (80B)
#define PLD 40    // Ps row stride f16 (80B)

__global__ __launch_bounds__(256) void flash_attn(
    const _Float16* __restrict__ qf, const _Float16* __restrict__ kf,
    const _Float16* __restrict__ vf, float* __restrict__ ctx)
{
  __shared__ __align__(16) _Float16 Ks[2][32 * KLD];
  __shared__ __align__(16) _Float16 Vs[2][HN * VLD];
  __shared__ __align__(16) _Float16 Ps[8 * 16 * PLD];

  const int tid  = threadIdx.x;
  const int lane = tid & 31;
  const int wave = tid >> 5;
  const int hlf  = lane >> 4;
  const int l16  = lane & 15;

  const int qb = (blockIdx.x & 15) * 128;   // S/128 = 16 query blocks
  const int bn = blockIdx.x >> 4;           // b*NHEAD + n
  const _Float16* Q  = qf + (size_t)bn * S_LEN * HN;
  const _Float16* Kp = kf + (size_t)bn * S_LEN * HN;
  const _Float16* Vp = vf + (size_t)bn * S_LEN * HN;

  // Q fragments held in registers for the whole kernel (16 rows x 128)
  const int qrow = qb + wave * 16 + l16;
  v16h qa[4];
#pragma unroll
  for (int c = 0; c < 4; ++c) {
    const _Float16* p = Q + (size_t)qrow * HN + c * 32;
    qa[c] = cat8(*(const v8h*)(p + hlf * 16), *(const v8h*)(p + hlf * 16 + 8));
  }

  // staging work distribution
  const int kr  = tid >> 3;           // K: key row 0..31
  const int ksg = (tid & 7) * 16;     // K: 16-f16 segment
  const int vr  = tid >> 4;           // V: key pair 0..15
  const int vh0 = (tid & 15) * 8;     // V: hn group of 8
  v8h va, vb;

  auto stageK = [&](int k0, _Float16* Kbuf) {
    const _Float16* src = Kp + (size_t)(k0 + kr) * HN + ksg;
#ifdef HAVE_ASYNC_LDS
    async_cp16(src,     &Kbuf[kr * KLD + ksg]);
    async_cp16(src + 8, &Kbuf[kr * KLD + ksg + 8]);
#else
    *(v8h*)&Kbuf[kr * KLD + ksg]     = *(const v8h*)(src);
    *(v8h*)&Kbuf[kr * KLD + ksg + 8] = *(const v8h*)(src + 8);
#endif
  };
  auto loadV = [&](int k0) {
    va = *(const v8h*)(Vp + (size_t)(k0 + 2 * vr) * HN + vh0);
    vb = *(const v8h*)(Vp + (size_t)(k0 + 2 * vr + 1) * HN + vh0);
  };
  auto storeV = [&](_Float16* Vbuf) {   // transposed, key pairs packed 4B
#pragma unroll
    for (int j = 0; j < 8; ++j) {
      v2h pr; pr.x = va[j]; pr.y = vb[j];
      *(v2h*)&Vbuf[(vh0 + j) * VLD + 2 * vr] = pr;
    }
  };

  v8f zero = {0,0,0,0,0,0,0,0};
  v8f o[8];
#pragma unroll
  for (int j = 0; j < 8; ++j) o[j] = zero;
  float mrow[8], lrow[8];
#pragma unroll
  for (int r = 0; r < 8; ++r) { mrow[r] = -1e30f; lrow[r] = 0.0f; }

  const float scale = 0.08838834764831845f;  // 1/sqrt(128)
  const int nkb = (qb + 128) / 32;           // causal: key blocks needed

  stageK(0, Ks[0]);
  loadV(0);
  storeV(Vs[0]);
#ifdef HAVE_ASYNC_LDS
  wait_async0();
#endif
  __syncthreads();

  int cur = 0;
  for (int kb = 0; kb < nkb; ++kb) {
    const int k0 = kb * 32;
    const bool more = (kb + 1) < nkb;
    if (more) {
      stageK(k0 + 32, Ks[cur ^ 1]);
      loadV(k0 + 32);
    }

    // Scores: two 16-key tiles, contraction over HN in 4 chunks of 32
    v8f sc[2];
#pragma unroll
    for (int nt = 0; nt < 2; ++nt) {
      sc[nt] = zero;
#pragma unroll
      for (int c = 0; c < 4; ++c) {
        const _Float16* p = &Ks[cur][(nt * 16 + l16) * KLD + c * 32 + hlf * 16];
        sc[nt] = wmma16(qa[c], cat8(*(const v8h*)(p), *(const v8h*)(p + 8)), sc[nt]);
      }
    }

    // Scale + causal mask (rows = r + 8*half, cols = lane%16)
#pragma unroll
    for (int nt = 0; nt < 2; ++nt)
#pragma unroll
      for (int r = 0; r < 8; ++r) {
        float x = sc[nt][r] * scale;
        int key = k0 + nt * 16 + l16;
        int qr  = qb + wave * 16 + r + 8 * hlf;
        if (key > qr) x += -1.0e9f;
        sc[nt][r] = x;
      }

    // Online softmax
#pragma unroll
    for (int r = 0; r < 8; ++r) {
      float mx = fmaxf(sc[0][r], sc[1][r]);
#pragma unroll
      for (int d = 1; d < 16; d <<= 1) mx = fmaxf(mx, __shfl_xor(mx, d, 32));
      float mnew = fmaxf(mrow[r], mx);
      float sf = __expf(mrow[r] - mnew);
      mrow[r] = mnew;
      float p0 = __expf(sc[0][r] - mnew);
      float p1 = __expf(sc[1][r] - mnew);
      sc[0][r] = p0; sc[1][r] = p1;
      float rs = p0 + p1;
#pragma unroll
      for (int d = 1; d < 16; d <<= 1) rs += __shfl_xor(rs, d, 32);
      lrow[r] = lrow[r] * sf + rs;
#pragma unroll
      for (int j = 0; j < 8; ++j) o[j][r] *= sf;
    }

    // P: C-layout -> per-wave LDS -> A-fragment
#pragma unroll
    for (int nt = 0; nt < 2; ++nt)
#pragma unroll
      for (int r = 0; r < 8; ++r)
        Ps[(wave * 16 + r + 8 * hlf) * PLD + nt * 16 + l16] = (_Float16)sc[nt][r];

    v16h pa;
    {
      const _Float16* p = &Ps[(wave * 16 + l16) * PLD];
      pa = cat8(*(const v8h*)(p + hlf * 16), *(const v8h*)(p + hlf * 16 + 8));
    }

    // O += P @ V  (8 column tiles of 16, contraction over 32 keys)
#pragma unroll
    for (int j = 0; j < 8; ++j) {
      const _Float16* p = &Vs[cur][(j * 16 + l16) * VLD + hlf * 16];
      o[j] = wmma16(pa, cat8(*(const v8h*)(p), *(const v8h*)(p + 8)), o[j]);
    }

    if (more) {
      storeV(Vs[cur ^ 1]);
#ifdef HAVE_ASYNC_LDS
      wait_async0();
#endif
      cur ^= 1;
      __syncthreads();
    }
  }

  // Write ctx fp32 in (S,B,H) layout
  const int b = bn >> 4, n = bn & 15;
#pragma unroll
  for (int r = 0; r < 8; ++r) {
    float inv = 1.0f / lrow[r];
    int srow = qb + wave * 16 + r + 8 * hlf;
    float* dst = ctx + ((size_t)srow * BATCH + b) * HID + n * HN;
#pragma unroll
    for (int j = 0; j < 8; ++j) dst[j * 16 + l16] = o[j][r] * inv;
  }
}

// ---------------------------------------------------------------------------
extern "C" void kernel_launch(void* const* d_in, const int* in_sizes, int n_in,
                              void* d_out, int out_size, void* d_ws, size_t ws_size,
                              hipStream_t stream) {
  const float* hidden = (const float*)d_in[0];
  // d_in[1] attention_mask: causal -1e9 band, applied analytically in-kernel
  const float* Wqkv   = (const float*)d_in[2];
  const float* bqkv   = (const float*)d_in[3];
  const float* Wd     = (const float*)d_in[4];
  const float* bd     = (const float*)d_in[5];
  float* out = (float*)d_out;

  // Workspace layout
  char* ws = (char*)d_ws;
  float*    mixed = (float*)ws;                                  // 4096*6144 f32
  _Float16* qh = (_Float16*)(ws + (size_t)MROWS * 3 * HID * 4);
  _Float16* kh = qh + (size_t)BATCH * NHEAD * S_LEN * HN;
  _Float16* vh = kh + (size_t)BATCH * NHEAD * S_LEN * HN;
  float*    ctxp = (float*)(vh + (size_t)BATCH * NHEAD * S_LEN * HN);

  // 1) QKV projection: mixed = hidden @ Wqkv^T + bqkv
  dim3 g1(3 * HID / 128, MROWS / 256);
  gemm_bias_wmma<<<g1, 256, 0, stream>>>(hidden, Wqkv, bqkv, mixed,
                                         MROWS, 3 * HID, HID);

  // 2) RoPE + split to f16 [b][n][s][hn]
  int rope_threads = S_LEN * BATCH * NHEAD * 64;
  rope_split<<<rope_threads / 256, 256, 0, stream>>>(mixed, qh, kh, vh);

  // 3) Flash attention -> ctx (S,B,H) fp32
  flash_attn<<<BATCH * NHEAD * (S_LEN / 128), 256, 0, stream>>>(qh, kh, vh, ctxp);

  // 4) Dense projection: out = ctx @ Wd^T + bd
  dim3 g2(HID / 128, MROWS / 256);
  gemm_bias_wmma<<<g2, 256, 0, stream>>>(ctxp, Wd, bd, out,
                                         MROWS, HID, HID);
}
